// MultiHeadAttention_42674795053763
// MI455X (gfx1250) — matrix operand order repaired and checked by reading, not compiled, not run on hardware
//
#include <hip/hip_runtime.h>

typedef __bf16 bf16;
typedef __attribute__((ext_vector_type(16))) __bf16 v16bf;
typedef __attribute__((ext_vector_type(8)))  __bf16 v8bf;
typedef __attribute__((ext_vector_type(4)))  __bf16 v4bf;
typedef __attribute__((ext_vector_type(8)))  float  v8f;

#define DEV static __device__ __forceinline__

constexpr int Bsz   = 4;
constexpr int S     = 2048;
constexpr int Din   = 1024;         // input dim (also output dim)
constexpr int Hh    = 16;           // heads
constexpr int Dh    = 64;           // per-head dim
constexpr int Mrows = Bsz * S;      // 8192 global rows
constexpr int Htot  = Hh * Dh;      // 1024

struct U32x8 { uint4 lo, hi; };

// ---- fragment loaders -------------------------------------------------------
// A fragment (16x32 bf16, MxK): lane&15 = row; lanes 0-15 hold K {0..7,16..23},
// lanes 16-31 hold K {8..15,24..31} (ISA 7.12.2, 16-bit A 16x32).
DEV v16bf load_a_bf16(const bf16* __restrict__ base, int ld, int row0, int kb, int lane) {
  const int r = lane & 15, half = lane >> 4;
  const bf16* p = base + (size_t)(row0 + r) * ld + kb + half * 8;
  U32x8 s;
  s.lo = *reinterpret_cast<const uint4*>(p);        // K  kb+off .. +7
  s.hi = *reinterpret_cast<const uint4*>(p + 16);   // K  kb+off+16 .. +23
  return __builtin_bit_cast(v16bf, s);
}

// B fragment (32x16 bf16, KxN): lane&15 = column; lanes 0-15 hold K {0..7,16..23},
// lanes 16-31 hold K {8..15,24..31} (same K order as the A fragment).
// Source must be the transpose of B (row = column index), row-major.
DEV v16bf load_b_bf16(const bf16* __restrict__ base, int ld, int row0, int kb, int lane) {
  const int nl = lane & 15, half = lane >> 4;
  const bf16* p = base + (size_t)(row0 + nl) * ld + kb + half * 8;
  U32x8 s;
  s.lo = *reinterpret_cast<const uint4*>(p);
  s.hi = *reinterpret_cast<const uint4*>(p + 16);
  return __builtin_bit_cast(v16bf, s);
}

DEV v8f wmma_bf16(v16bf a, v16bf b, v8f c) {
  return __builtin_amdgcn_wmma_f32_16x16x32_bf16(false, a, false, b, (short)0, c,
                                                 false, false);
}

// ---- activation f32 -> bf16 streaming convert ------------------------------
__global__ __launch_bounds__(256) void xcvt_kernel(const float* __restrict__ in,
                                                   bf16* __restrict__ out) {
  const size_t idx = ((size_t)blockIdx.x * blockDim.x + threadIdx.x) * 4;
  float4 f = *reinterpret_cast<const float4*>(in + idx);
  v4bf o;
  o[0] = (bf16)f.x; o[1] = (bf16)f.y; o[2] = (bf16)f.z; o[3] = (bf16)f.w;
  *reinterpret_cast<v4bf*>(out + idx) = o;
}

// ---- weight transpose + f32->bf16: Wt[n][k] = W[k][n] ----------------------
__global__ __launch_bounds__(256) void wtrans_kernel(const float* __restrict__ W,
                                                     bf16* __restrict__ Wt) {
  int idx = blockIdx.x * blockDim.x + threadIdx.x;   // over 1024*1024
  int n = idx >> 10;
  int k = idx & 1023;
  Wt[idx] = (bf16)W[(size_t)k * Din + n];            // writes coalesced
}

// ---- stage 1: projections  C = Xb @ W + b  (bf16 x bf16 -> f32) ------------
// 32x32 tile per wave: 2 A frags x 2 B frags -> 4 WMMA per k-step.
// TSTORE==0: row-major bf16 [Mrows x Htot]   (Q)
// TSTORE==1: per-head transposed bf16 [B][Htot][S] (K, V) with packed stores
template <int TSTORE>
__global__ __launch_bounds__(256) void proj_kernel(const bf16*  __restrict__ Xb,
                                                   const bf16*  __restrict__ Wt,
                                                   const float* __restrict__ bias,
                                                   bf16* __restrict__ out) {
  const int lane = threadIdx.x & 31;
  const int wave = threadIdx.x >> 5;
  const int job  = blockIdx.x * 8 + wave;   // 8192 jobs: 256 mtiles * 32 ntiles
  const int m0   = (job >> 5) * 32;
  const int n0   = (job & 31) * 32;

  v8f acc[2][2] = {{{}, {}}, {{}, {}}};
  for (int kb = 0; kb < Din; kb += 32) {
    v16bf a0 = load_a_bf16(Xb, Din, m0,      kb, lane);
    v16bf a1 = load_a_bf16(Xb, Din, m0 + 16, kb, lane);
    v16bf b0 = load_b_bf16(Wt, Din, n0,      kb, lane);
    v16bf b1 = load_b_bf16(Wt, Din, n0 + 16, kb, lane);
    acc[0][0] = wmma_bf16(a0, b0, acc[0][0]);
    acc[0][1] = wmma_bf16(a0, b1, acc[0][1]);
    acc[1][0] = wmma_bf16(a1, b0, acc[1][0]);
    acc[1][1] = wmma_bf16(a1, b1, acc[1][1]);
  }
  const int nl = lane & 15, half = lane >> 4;
  const float bias0 = bias[n0 + nl];
  const float bias1 = bias[n0 + 16 + nl];

#pragma unroll
  for (int mi = 0; mi < 2; ++mi) {
    const int mBase = m0 + mi * 16;
    if (TSTORE == 0) {
#pragma unroll
      for (int r = 0; r < 8; ++r) {
        int m = mBase + r + 8 * half;                  // C: VGPR r -> row r(+8)
        out[(size_t)m * Htot + n0 + nl]      = (bf16)(acc[mi][0][r] + bias0);
        out[(size_t)m * Htot + n0 + 16 + nl] = (bf16)(acc[mi][1][r] + bias1);
      }
    } else {
      // out[((b*Hh+h)*Dh+dd)*S + s]; 8 consecutive s per lane -> one 16B store
      const int b  = mBase >> 11;
      const int s0 = (mBase & 2047) + 8 * half;
      v8bf v0, v1;
#pragma unroll
      for (int r = 0; r < 8; ++r) {
        v0[r] = (bf16)(acc[mi][0][r] + bias0);
        v1[r] = (bf16)(acc[mi][1][r] + bias1);
      }
      bf16* pbase = out + (size_t)b * ((size_t)Htot * S);
      *reinterpret_cast<v8bf*>(pbase + (size_t)(n0 + nl)      * S + s0) = v0;
      *reinterpret_cast<v8bf*>(pbase + (size_t)(n0 + 16 + nl) * S + s0) = v1;
    }
  }
}

// ---- stage 2: per head  Mt[j][i] = scale * sum_s V[s][j] * K[s][i] ---------
// A = Vt rows (contiguous), B = Kt rows (contiguous). 64x64 out per (b,h).
__global__ __launch_bounds__(512) void kv_outer_kernel(const bf16* __restrict__ Kt,
                                                       const bf16* __restrict__ Vt,
                                                       bf16* __restrict__ Mt) {
  const int lane = threadIdx.x & 31;
  const int wave = threadIdx.x >> 5;        // 0..15
  const int bh   = blockIdx.x;              // 0..63
  const int j0   = (wave >> 2) * 16;        // Mt row tile
  const int i0   = (wave & 3) * 16;         // Mt col tile
  const bf16* Vb = Vt + (size_t)bh * Dh * S;
  const bf16* Kb = Kt + (size_t)bh * Dh * S;

  v8f acc = {};
  for (int kb = 0; kb < S; kb += 32) {
    v16bf a = load_a_bf16(Vb, S, j0, kb, lane);
    v16bf b = load_b_bf16(Kb, S, i0, kb, lane);
    acc = wmma_bf16(a, b, acc);
  }
  const float scale = 0.125f;               // 1/sqrt(64)
  const int nl = lane & 15, half = lane >> 4;
  bf16* outp = Mt + (size_t)bh * (Dh * Dh);
#pragma unroll
  for (int r = 0; r < 8; ++r)
    outp[(j0 + r + 8 * half) * Dh + i0 + nl] = (bf16)(acc[r] * scale);
}

// ---- stage 3: per head  Abuf[s][h*64+j] = sum_i Q[s][i] * M[i][j] ----------
// 32-row tile x full head width: 2 A frags x 4 B frags -> 8 WMMA per k-step.
__global__ __launch_bounds__(256) void attn_kernel(const bf16* __restrict__ Qn,
                                                   const bf16* __restrict__ Mt,
                                                   bf16* __restrict__ Abuf) {
  const int lane = threadIdx.x & 31;
  const int wave = threadIdx.x >> 5;
  const int job  = blockIdx.x * 8 + wave;   // 4096 jobs: 256 mtiles * 16 heads
  const int m0   = (job >> 4) * 32;
  const int h    = job & 15;
  const int b    = m0 >> 11;
  const bf16* Mb = Mt + (size_t)(b * Hh + h) * (Dh * Dh);

  v8f acc[2][4] = {{{}, {}, {}, {}}, {{}, {}, {}, {}}};
  for (int kb = 0; kb < Dh; kb += 32) {
    v16bf a0 = load_a_bf16(Qn, Htot, m0,      h * Dh + kb, lane);
    v16bf a1 = load_a_bf16(Qn, Htot, m0 + 16, h * Dh + kb, lane);
#pragma unroll
    for (int t = 0; t < 4; ++t) {
      v16bf bb = load_b_bf16(Mb, Dh, t * 16, kb, lane);
      acc[0][t] = wmma_bf16(a0, bb, acc[0][t]);
      acc[1][t] = wmma_bf16(a1, bb, acc[1][t]);
    }
  }
  const int nl = lane & 15, half = lane >> 4;
#pragma unroll
  for (int mi = 0; mi < 2; ++mi)
#pragma unroll
    for (int t = 0; t < 4; ++t)
#pragma unroll
      for (int r = 0; r < 8; ++r)
        Abuf[(size_t)(m0 + mi * 16 + r + 8 * half) * Htot + h * Dh + t * 16 + nl] =
            (bf16)acc[mi][t][r];
}

// ---- stage 4: out = Abuf @ Wo + bo  (f32 output), 32x32 wave tile ----------
__global__ __launch_bounds__(256) void out_kernel(const bf16* __restrict__ Abuf,
                                                  const bf16* __restrict__ Wot,
                                                  const float* __restrict__ bo,
                                                  float* __restrict__ outp) {
  const int lane = threadIdx.x & 31;
  const int wave = threadIdx.x >> 5;
  const int job  = blockIdx.x * 8 + wave;   // 8192 jobs
  const int m0   = (job >> 5) * 32;
  const int n0   = (job & 31) * 32;

  v8f acc[2][2] = {{{}, {}}, {{}, {}}};
  for (int kb = 0; kb < Htot; kb += 32) {
    v16bf a0 = load_a_bf16(Abuf, Htot, m0,      kb, lane);
    v16bf a1 = load_a_bf16(Abuf, Htot, m0 + 16, kb, lane);
    v16bf b0 = load_b_bf16(Wot, Htot, n0,      kb, lane);
    v16bf b1 = load_b_bf16(Wot, Htot, n0 + 16, kb, lane);
    acc[0][0] = wmma_bf16(a0, b0, acc[0][0]);
    acc[0][1] = wmma_bf16(a0, b1, acc[0][1]);
    acc[1][0] = wmma_bf16(a1, b0, acc[1][0]);
    acc[1][1] = wmma_bf16(a1, b1, acc[1][1]);
  }
  const int nl = lane & 15, half = lane >> 4;
  const float bias0 = bo[n0 + nl], bias1 = bo[n0 + 16 + nl];
#pragma unroll
  for (int mi = 0; mi < 2; ++mi)
#pragma unroll
    for (int r = 0; r < 8; ++r) {
      int m = m0 + mi * 16 + r + 8 * half;
      outp[(size_t)m * Din + n0 + nl]      = acc[mi][0][r] + bias0;
      outp[(size_t)m * Din + n0 + 16 + nl] = acc[mi][1][r] + bias1;
    }
}

// ---- host ------------------------------------------------------------------
extern "C" void kernel_launch(void* const* d_in, const int* in_sizes, int n_in,
                              void* d_out, int out_size, void* d_ws, size_t ws_size,
                              hipStream_t stream) {
  (void)in_sizes; (void)n_in; (void)out_size; (void)ws_size;
  const float* q  = (const float*)d_in[0];
  const float* k  = (const float*)d_in[1];
  const float* v  = (const float*)d_in[2];
  const float* Wq = (const float*)d_in[3];
  const float* bq = (const float*)d_in[4];
  const float* Wk = (const float*)d_in[5];
  const float* bk = (const float*)d_in[6];
  const float* Wv = (const float*)d_in[7];
  const float* bv = (const float*)d_in[8];
  const float* Wo = (const float*)d_in[9];
  const float* bo = (const float*)d_in[10];
  float* outp = (float*)d_out;

  char* ws = (char*)d_ws;
  size_t off = 0;
  auto take = [&](size_t bytes) {
    char* p = ws + off;
    off += (bytes + 255) & ~(size_t)255;
    return p;
  };
  const size_t wbytes = (size_t)Din * Htot * sizeof(bf16);     // 2 MB
  const size_t abytes = (size_t)Mrows * Htot * sizeof(bf16);   // 16 MB
  bf16* Wqt = (bf16*)take(wbytes);
  bf16* Wkt = (bf16*)take(wbytes);
  bf16* Wvt = (bf16*)take(wbytes);
  bf16* Wot = (bf16*)take(wbytes);
  bf16* Qn  = (bf16*)take(abytes);
  bf16* Kt  = (bf16*)take(abytes);
  bf16* Vt  = (bf16*)take(abytes);
  bf16* Xb  = (bf16*)take(abytes);   // staging: bf16 copy of current activation
  bf16* Mt  = (bf16*)take((size_t)Bsz * Hh * Dh * Dh * sizeof(bf16)); // 512 KB
  bf16* Abuf = Kt;  // Kt is dead after stage 2 -> reuse for stage-3 output

  // weights -> transposed bf16
  wtrans_kernel<<<(Din * Htot) / 256, 256, 0, stream>>>(Wq, Wqt);
  wtrans_kernel<<<(Din * Htot) / 256, 256, 0, stream>>>(Wk, Wkt);
  wtrans_kernel<<<(Din * Htot) / 256, 256, 0, stream>>>(Wv, Wvt);
  wtrans_kernel<<<(Din * Htot) / 256, 256, 0, stream>>>(Wo, Wot);

  const int cvtBlocks  = (Mrows * Din) / (256 * 4);   // 8192
  const int gemmBlocks = 1024;                        // 8192 jobs / 8 waves

  // projections (activation converted once, staged in Xb, reused per GEMM)
  xcvt_kernel<<<cvtBlocks, 256, 0, stream>>>(q, Xb);
  proj_kernel<0><<<gemmBlocks, 256, 0, stream>>>(Xb, Wqt, bq, Qn);  // Q row-major
  xcvt_kernel<<<cvtBlocks, 256, 0, stream>>>(k, Xb);
  proj_kernel<1><<<gemmBlocks, 256, 0, stream>>>(Xb, Wkt, bk, Kt);  // K transposed
  xcvt_kernel<<<cvtBlocks, 256, 0, stream>>>(v, Xb);
  proj_kernel<1><<<gemmBlocks, 256, 0, stream>>>(Xb, Wvt, bv, Vt);  // V transposed

  // linear attention: M = scale * K^T V
  kv_outer_kernel<<<Bsz * Hh, 512, 0, stream>>>(Kt, Vt, Mt);

  // A = Q @ M
  attn_kernel<<<512, 256, 0, stream>>>(Qn, Mt, Abuf);

  // out = A @ Wo + bo
  out_kernel<<<gemmBlocks, 256, 0, stream>>>(Abuf, Wot, bo, outp);
}
